// MultiHeadAttn_27290222199184
// MI455X (gfx1250) — compile-verified, matrix operand order checked by reading, not run on hardware
//
#include <hip/hip_runtime.h>
#include <hip/hip_bf16.h>

// Transformer-XL relative multi-head attention, CDNA5 (gfx1250) WMMA pipeline.
// GEMMs use v_wmma_f32_16x16x32_f16 (fp32 accumulate). QKV GEMM stages weight
// tiles in LDS via gfx1250 async global->LDS copies (ASYNCcnt) double-buffered
// over K. Attention is fused flash-style per (batch, head, 16-row query tile);
// rel_shift realized via sliding-window QR=qr*rk^T WMMA + LDS diagonal gather.

#define T_    1024
#define MEMW  1024
#define BW    4
#define DM_   1024
#define NH    16
#define HD    64
#define CW    2048
#define NDW   1024
#define SCALE_ 0.125f   // 1/sqrt(64)

typedef __attribute__((ext_vector_type(16))) _Float16     v16h;
typedef __attribute__((ext_vector_type(8)))  float        v8f;
typedef __attribute__((ext_vector_type(8)))  unsigned int v8u;

static __device__ inline unsigned int pkf16(float a, float b) {
  union { _Float16 h[2]; unsigned int u; } x;
  x.h[0] = (_Float16)a; x.h[1] = (_Float16)b;
  return x.u;
}

static __device__ inline v16h as_v16h(v8u u) {
  union { v8u u; v16h h; } x; x.u = u; return x.h;
}

// K offset (even; pair {K,K+1}) held in dword v (0..7) of a lane for the
// 16x32 f16 WMMA A/B fragment layout (ISA 7.12.2): VGPR0 K=0,1 / K=8,9 ...
static __device__ inline int kmap(int v, int hi) {
  return 2 * (v & 3) + 8 * hi + 16 * (v >> 2);
}

// 16x32 fragment from row-major f16 memory; one (row,Kpair) dword per slot.
static __device__ inline v16h ld_f16_frag(const unsigned int* dw, int row,
                                          int stride_dw, int kofs, int hi) {
  v8u u;
#pragma unroll
  for (int v = 0; v < 8; ++v)
    u[v] = dw[(size_t)row * stride_dw + ((kofs + kmap(v, hi)) >> 1)];
  return as_v16h(u);
}

// 16x32 A-fragment from a row-major f32 row pointer (convert to f16).
static __device__ inline v16h ld_f32_row_frag(const float* rowptr, int kofs, int hi) {
  v8u u;
#pragma unroll
  for (int v = 0; v < 8; ++v) {
    int K = kofs + kmap(v, hi);
    float2 f = *(const float2*)(rowptr + K);
    u[v] = pkf16(f.x, f.y);
  }
  return as_v16h(u);
}

// 16x32 B-fragment: element (K,N) from f32 W[(kofs+K)*ldw + col].
static __device__ inline v16h ld_f32_col_frag(const float* W, int ldw, int kofs,
                                              int col, int hi) {
  v8u u;
#pragma unroll
  for (int v = 0; v < 8; ++v) {
    int K = kofs + kmap(v, hi);
    u[v] = pkf16(W[(size_t)K * ldw + col], W[(size_t)(K + 1) * ldw + col]);
  }
  return as_v16h(u);
}

static __device__ inline v8f wmma16(v16h a, v16h b, v8f c) {
  return __builtin_amdgcn_wmma_f32_16x16x32_f16(false, a, false, b, (short)0, c,
                                                false, false);
}

// gfx1250 async global->LDS copy (16 bytes per lane), tracked by ASYNCcnt.
static __device__ inline void async_g2l_b128(unsigned lds_byte_off, const void* gaddr) {
  asm volatile("global_load_async_to_lds_b128 %0, %1, off"
               :: "v"(lds_byte_off), "v"(gaddr) : "memory");
}

// ---------------------------------------------------------------------------
// Kernel 1: w_heads = cat(m,h) @ W_qkv ; split into q(+biases)/k/v, f16 stage.
// q -> qw/qr [B,N,T,D]; k -> [B,N,C,D]; v -> transposed [B,N,D,C].
// B-tile (128K x 16 cols f32) staged in LDS with async copies, double-buffered;
// all 8 waves of a block share the same column tile.
// ---------------------------------------------------------------------------
__launch_bounds__(256)
__global__ void qkv_kernel(const float* __restrict__ h, const float* __restrict__ m,
                           const float* __restrict__ Wqkv,
                           const float* __restrict__ rwb, const float* __restrict__ rrb,
                           _Float16* __restrict__ qw, _Float16* __restrict__ qr,
                           _Float16* __restrict__ kx, _Float16* __restrict__ vT) {
  const int lane = threadIdx.x & 31, wid = threadIdx.x >> 5;
  const int hi = lane >> 4, ln16 = lane & 15;
  const int gw = blockIdx.x * 8 + wid;       // 98304 wave-tiles
  const int mt = gw & 511, nt = gw >> 9;     // M=8192/16, N=3072/16 (nt uniform/block)
  const int row = mt * 16 + ln16;            // cat row = c*4+b
  const float* arow = (row < MEMW * BW) ? (m + (size_t)row * DM_)
                                        : (h + (size_t)(row - MEMW * BW) * DM_);
  const int col0 = nt * 16;

  __shared__ float btile[2][128][16];        // K-chunk x 16 columns, double-buffered

  // Issue async copies of one 128x16 f32 W_qkv chunk into LDS buffer `buf`.
  auto stage = [&](int kbase, int buf) {
    for (int e = threadIdx.x; e < 512; e += 256) {      // 512 x 16B transfers
      int rr = e >> 2, cp = (e & 3) * 4;
      unsigned lds = (unsigned)(uintptr_t)&btile[buf][rr][cp];
      const float* g = Wqkv + (size_t)(kbase + rr) * (3 * NDW) + col0 + cp;
      async_g2l_b128(lds, g);
    }
  };

  stage(0, 0);
  v8f acc = {};
  for (int c = 0; c < 8; ++c) {              // 8 K-chunks of 128
    if (c < 7) {
      stage((c + 1) * 128, (c + 1) & 1);
      asm volatile("s_wait_asynccnt 0x2" ::: "memory");  // chunk c complete
    } else {
      asm volatile("s_wait_asynccnt 0x0" ::: "memory");
    }
    __syncthreads();
    const float* bt = &btile[c & 1][0][0];
#pragma unroll
    for (int ks = 0; ks < 4; ++ks) {
      v16h A = ld_f32_row_frag(arow, c * 128 + ks * 32, hi);
      v16h B = ld_f32_col_frag(bt, 16, ks * 32, ln16, hi);
      acc = wmma16(A, B, acc);
    }
    __syncthreads();                         // before next chunk overwrites buf
  }

  const int col = col0 + ln16;
#pragma unroll
  for (int v = 0; v < 8; ++v) {
    int rw = mt * 16 + v + 8 * hi;
    int cc = rw >> 2, bb = rw & 3;
    float val = acc[v];
    if (col < NDW) {                        // q: only last T rows
      if (cc >= MEMW) {
        int t = cc - MEMW, n = col >> 6, d = col & 63;
        size_t o = (((size_t)(bb * NH + n)) * T_ + t) * HD + d;
        qw[o] = (_Float16)(val + rwb[col]);
        qr[o] = (_Float16)(val + rrb[col]);
      }
    } else if (col < 2 * NDW) {             // k
      int c2 = col - NDW, n = c2 >> 6, d = c2 & 63;
      kx[(((size_t)(bb * NH + n)) * CW + cc) * HD + d] = (_Float16)val;
    } else {                                // v (transposed: [b,n,d,C])
      int c2 = col - 2 * NDW, n = c2 >> 6, d = c2 & 63;
      vT[(((size_t)(bb * NH + n)) * HD + d) * CW + cc] = (_Float16)val;
    }
  }
}

// ---------------------------------------------------------------------------
// Kernel 2: r_k = r @ W_r  -> f16 [N, C, D]
// ---------------------------------------------------------------------------
__launch_bounds__(256)
__global__ void rproj_kernel(const float* __restrict__ r, const float* __restrict__ Wr,
                             _Float16* __restrict__ rk) {
  const int lane = threadIdx.x & 31, wid = threadIdx.x >> 5;
  const int hi = lane >> 4, ln16 = lane & 15;
  const int gw = blockIdx.x * 8 + wid;       // 8192 wave-tiles
  const int mt = gw & 127, nt = gw >> 7;     // M=2048/16, N=1024/16
  const float* arow = r + (size_t)(mt * 16 + ln16) * DM_;
  const int col = nt * 16 + ln16;
  v8f acc = {};
  for (int kk = 0; kk < DM_; kk += 32) {
    v16h A = ld_f32_row_frag(arow, kk, hi);
    v16h B = ld_f32_col_frag(Wr, NDW, kk, col, hi);
    acc = wmma16(A, B, acc);
  }
#pragma unroll
  for (int v = 0; v < 8; ++v) {
    int rw = mt * 16 + v + 8 * hi;
    int n = col >> 6, d = col & 63;
    rk[((size_t)n * CW + rw) * HD + d] = (_Float16)acc[v];
  }
}

// ---------------------------------------------------------------------------
// Kernel 3: fused flash attention with Transformer-XL rel_shift.
// One wave = (b, head, 16-query-row tile). 8 independent waves per block.
// BD[i,j] = qr_i . rk[j - i + T - 1]  (valid region always in-range; masked
// positions clamped). Sliding window of three QR WMMA tiles -> LDS gather.
// ---------------------------------------------------------------------------
__launch_bounds__(256)
__global__ void attn_kernel(const _Float16* __restrict__ qw, const _Float16* __restrict__ qr,
                            const _Float16* __restrict__ kx, const _Float16* __restrict__ vT,
                            const _Float16* __restrict__ rkp, _Float16* __restrict__ vec) {
  const int lane = threadIdx.x & 31, wid = threadIdx.x >> 5;
  const int hi = lane >> 4, ln16 = lane & 15;
  const int gw = blockIdx.x * 8 + wid;                 // 4096 waves
  const int it = gw & 63, n = (gw >> 6) & 15, b = gw >> 10;
  const int i0 = it * 16;

  __shared__ float    qr_strip[8][16][50];             // 48-wide QR window (+pad)
  __shared__ _Float16 p_strip[8][16][32];              // P tile C->A relayout
  float    (*QS)[50] = qr_strip[wid];
  _Float16 (*PS)[32] = p_strip[wid];

  const unsigned int* qw_dw = (const unsigned int*)(qw + (size_t)(b * NH + n) * T_ * HD);
  const unsigned int* qr_dw = (const unsigned int*)(qr + (size_t)(b * NH + n) * T_ * HD);
  const unsigned int* k_dw  = (const unsigned int*)(kx + (size_t)(b * NH + n) * CW * HD);
  const unsigned int* v_dw  = (const unsigned int*)(vT + (size_t)(b * NH + n) * HD * CW);
  const unsigned int* rk_dw = (const unsigned int*)(rkp + (size_t)n * CW * HD);

  // Query A-fragments (16 x 64 -> two K=32 fragments each)
  const int tq = i0 + ln16;
  v16h qwA0 = ld_f16_frag(qw_dw, tq, HD / 2, 0,  hi);
  v16h qwA1 = ld_f16_frag(qw_dw, tq, HD / 2, 32, hi);
  v16h qrA0 = ld_f16_frag(qr_dw, tq, HD / 2, 0,  hi);
  v16h qrA1 = ld_f16_frag(qr_dw, tq, HD / 2, 32, hi);

  v8f O[4];
#pragma unroll
  for (int g = 0; g < 4; ++g) O[g] = (v8f){};
  float mrow[8], lrow[8];
#pragma unroll
  for (int v = 0; v < 8; ++v) { mrow[v] = -3.0e38f; lrow[v] = 0.f; }

  const int base0 = T_ - 1 - i0;
  auto qrtile = [&](int s0) -> v8f {
    int s = s0 + ln16;
    s = s < 0 ? 0 : (s > CW - 1 ? CW - 1 : s);        // clamped rows feed only masked scores
    v16h B0 = ld_f16_frag(rk_dw, s, HD / 2, 0,  hi);
    v16h B1 = ld_f16_frag(rk_dw, s, HD / 2, 32, hi);
    v8f a = {};
    a = wmma16(qrA0, B0, a);
    a = wmma16(qrA1, B1, a);
    return a;
  };

  v8f QRa = qrtile(base0 - 16), QRb, QRc;
  const int jend = i0 + 16 + MEMW;                     // causal bound (exclusive)

  for (int j0 = 0; j0 < jend; j0 += 32) {
    const int base = base0 + j0;

    // Prefetch next chunk's K rows and rk rows (global_prefetch_b8).
    if (j0 + 32 < jend) {
      __builtin_prefetch((const char*)(k_dw + (size_t)(j0 + 32 + lane) * (HD / 2)), 0, 1);
      int sp = base + 32 + lane; sp = sp > CW - 1 ? CW - 1 : sp;
      __builtin_prefetch((const char*)(rk_dw + (size_t)sp * (HD / 2)), 0, 1);
    }

    QRb = qrtile(base);
    QRc = qrtile(base + 16);

    // AC tiles for key columns [j0, j0+32)
    v8f AC0 = {}, AC1 = {};
    {
      int j = j0 + ln16;
      v16h B0 = ld_f16_frag(k_dw, j, HD / 2, 0,  hi);
      v16h B1 = ld_f16_frag(k_dw, j, HD / 2, 32, hi);
      AC0 = wmma16(qwA0, B0, AC0);
      AC0 = wmma16(qwA1, B1, AC0);
      j = j0 + 16 + ln16;
      B0 = ld_f16_frag(k_dw, j, HD / 2, 0,  hi);
      B1 = ld_f16_frag(k_dw, j, HD / 2, 32, hi);
      AC1 = wmma16(qwA0, B0, AC1);
      AC1 = wmma16(qwA1, B1, AC1);
    }

    // Spill QR window to LDS, gather rel-shifted BD (in-order per-wave LDS).
#pragma unroll
    for (int v = 0; v < 8; ++v) {
      int M = v + 8 * hi;
      QS[M][ln16]      = QRa[v];
      QS[M][16 + ln16] = QRb[v];
      QS[M][32 + ln16] = QRc[v];
    }

    float sc0[8], sc1[8];
#pragma unroll
    for (int v = 0; v < 8; ++v) {
      int di = v + 8 * hi;
      sc0[v] = (AC0[v] + QS[di][16 + ln16 - di]) * SCALE_;
      sc1[v] = (AC1[v] + QS[di][32 + ln16 - di]) * SCALE_;
      int lim = i0 + di + MEMW;                        // mask: j > i + MEM
      if (j0 + ln16 > lim)      sc0[v] = -65000.f;
      if (j0 + 16 + ln16 > lim) sc1[v] = -65000.f;
    }

    // Online softmax (row = M; columns live across the 16-lane group).
    float p0[8], p1[8];
#pragma unroll
    for (int v = 0; v < 8; ++v) {
      float tm = fmaxf(sc0[v], sc1[v]);
#pragma unroll
      for (int d = 1; d < 16; d <<= 1) tm = fmaxf(tm, __shfl_xor(tm, d));
      float mnew = fmaxf(mrow[v], tm);
      float corr = __expf(mrow[v] - mnew);
      mrow[v] = mnew;
      p0[v] = __expf(sc0[v] - mnew);
      p1[v] = __expf(sc1[v] - mnew);
      float rs = p0[v] + p1[v];
#pragma unroll
      for (int d = 1; d < 16; d <<= 1) rs += __shfl_xor(rs, d);
      lrow[v] = lrow[v] * corr + rs;
#pragma unroll
      for (int g = 0; g < 4; ++g) O[g][v] *= corr;
    }

    // P (C-layout) -> LDS -> f16 A-fragment (16x32).
#pragma unroll
    for (int v = 0; v < 8; ++v) {
      int M = v + 8 * hi;
      PS[M][ln16]      = (_Float16)p0[v];
      PS[M][16 + ln16] = (_Float16)p1[v];
    }
    v8u pu;
#pragma unroll
    for (int v = 0; v < 8; ++v)
      pu[v] = *(const unsigned int*)&PS[ln16][kmap(v, hi)];
    v16h Pa = as_v16h(pu);

    // O += P @ V  (V transposed so K runs contiguously).
#pragma unroll
    for (int g = 0; g < 4; ++g) {
      v16h Vb = ld_f16_frag(v_dw, g * 16 + ln16, CW / 2, j0, hi);
      O[g] = wmma16(Pa, Vb, O[g]);
    }

    QRa = QRc;                                         // slide window
  }

  // Normalize and store vec as f16 [T, B, N*D].
#pragma unroll
  for (int v = 0; v < 8; ++v) {
    float inv = 1.0f / lrow[v];
    int i = i0 + v + 8 * hi;
    size_t rbase = ((size_t)i * BW + b) * NDW + n * HD;
#pragma unroll
    for (int g = 0; g < 4; ++g)
      vec[rbase + g * 16 + ln16] = (_Float16)(O[g][v] * inv);
  }
}

// ---------------------------------------------------------------------------
// Kernel 4: x = h + vec @ W_o    (f32 result to workspace)
// ---------------------------------------------------------------------------
__launch_bounds__(256)
__global__ void oproj_kernel(const _Float16* __restrict__ vec, const float* __restrict__ Wo,
                             const float* __restrict__ h, float* __restrict__ xbuf) {
  const int lane = threadIdx.x & 31, wid = threadIdx.x >> 5;
  const int hi = lane >> 4, ln16 = lane & 15;
  const int gw = blockIdx.x * 8 + wid;       // 16384 wave-tiles
  const int mt = gw & 255, nt = gw >> 8;     // M=4096/16, N=1024/16
  const unsigned int* vdw = (const unsigned int*)vec;
  const int row = mt * 16 + ln16;
  const int col = nt * 16 + ln16;
  v8f acc = {};
  for (int kk = 0; kk < NDW; kk += 32) {
    v16h A = ld_f16_frag(vdw, row, NDW / 2, kk, hi);
    v16h B = ld_f32_col_frag(Wo, DM_, kk, col, hi);
    acc = wmma16(A, B, acc);
  }
#pragma unroll
  for (int v = 0; v < 8; ++v) {
    int rw = mt * 16 + v + 8 * hi;
    xbuf[(size_t)rw * DM_ + col] = acc[v] + h[(size_t)rw * DM_ + col];
  }
}

// ---------------------------------------------------------------------------
// Kernel 5: LayerNorm over DM per row.
// ---------------------------------------------------------------------------
__launch_bounds__(256)
__global__ void ln_kernel(const float* __restrict__ x, const float* __restrict__ g,
                          const float* __restrict__ bt, float* __restrict__ out) {
  const int row = blockIdx.x;
  const float* xr = x + (size_t)row * DM_;
  float s1 = 0.f, s2 = 0.f;
  for (int c = threadIdx.x; c < DM_; c += 256) {
    float v = xr[c];
    s1 += v; s2 += v * v;
  }
#pragma unroll
  for (int d = 1; d < 32; d <<= 1) { s1 += __shfl_xor(s1, d); s2 += __shfl_xor(s2, d); }
  __shared__ float sm1[8], sm2[8];
  if ((threadIdx.x & 31) == 0) { sm1[threadIdx.x >> 5] = s1; sm2[threadIdx.x >> 5] = s2; }
  __syncthreads();
  float t1 = 0.f, t2 = 0.f;
#pragma unroll
  for (int w = 0; w < 8; ++w) { t1 += sm1[w]; t2 += sm2[w]; }
  float mu  = t1 * (1.0f / DM_);
  float var = t2 * (1.0f / DM_) - mu * mu;
  float rstd = rsqrtf(var + 1e-5f);
  for (int c = threadIdx.x; c < DM_; c += 256)
    out[(size_t)row * DM_ + c] = (xr[c] - mu) * rstd * g[c] + bt[c];
}

// ---------------------------------------------------------------------------
extern "C" void kernel_launch(void* const* d_in, const int* in_sizes, int n_in,
                              void* d_out, int out_size, void* d_ws, size_t ws_size,
                              hipStream_t stream) {
  const float* h     = (const float*)d_in[0];
  const float* m     = (const float*)d_in[1];
  const float* r     = (const float*)d_in[2];
  // d_in[3]: mask (bool) — computed analytically (j > i + MEM), unused.
  const float* Wqkv  = (const float*)d_in[4];
  const float* Wr    = (const float*)d_in[5];
  const float* Wo    = (const float*)d_in[6];
  const float* rwb   = (const float*)d_in[7];
  const float* rrb   = (const float*)d_in[8];
  const float* gamma = (const float*)d_in[9];
  const float* beta  = (const float*)d_in[10];
  float* out = (float*)d_out;

  // Workspace carve-up (~76 MB total).
  _Float16* qw  = (_Float16*)d_ws;                  // [B,N,T,D]   4M halves
  _Float16* qr  = qw  + (size_t)4 * 1024 * 1024;    // [B,N,T,D]   4M
  _Float16* kx  = qr  + (size_t)4 * 1024 * 1024;    // [B,N,C,D]   8M
  _Float16* vT  = kx  + (size_t)8 * 1024 * 1024;    // [B,N,D,C]   8M
  _Float16* rk  = vT  + (size_t)8 * 1024 * 1024;    // [N,C,D]     2M
  _Float16* vec = rk  + (size_t)2 * 1024 * 1024;    // [T,B,N*D]   4M
  float*    xbuf = (float*)(vec + (size_t)4 * 1024 * 1024);  // [T*B, DM] f32

  qkv_kernel <<<12288, 256, 0, stream>>>(h, m, Wqkv, rwb, rrb, qw, qr, kx, vT);
  rproj_kernel<<<1024,  256, 0, stream>>>(r, Wr, rk);
  attn_kernel <<<512,   256, 0, stream>>>(qw, qr, kx, vT, rk, vec);
  oproj_kernel<<<2048,  256, 0, stream>>>(vec, Wo, h, xbuf);
  ln_kernel   <<<4096,  256, 0, stream>>>(xbuf, gamma, beta, out);
}
